// Attention_14525579395337
// MI455X (gfx1250) — matrix operand order repaired and checked by reading, not compiled, not run on hardware
//
#include <hip/hip_runtime.h>
#include <hip/hip_bf16.h>

// ---------------------------------------------------------------------------
// Bahdanau additive attention with coverage, CDNA5 (gfx1250, wave32).
//   B=8, T=32, L=512, D=1024
// Pipeline:
//   k0: bf16 copies of query/states (bandwidth pass; A-operands of GEMMs)
//   k1: qf  = query @ Wq + bq                     (GEMM 256x1024, K=1024)
//   k2: sc  = states @ Ws + coverage x Wcov       (GEMM 4096x1024, K=1024)
//   k3: align = v . tanh(qf+sc); softmax; attn_t / new_coverage; attn -> bf16
//   k4: ctx = attn @ states                       (batched GEMM 32x1024, K=512)
//   k5: h1  = relu([ctx,query] @ W1 + b1)         (split-A GEMM, K=2048)
//   k6: h   = h1 @ W2 + b2                        (GEMM 256x1024, K=1024)
// GEMM: 128x64 tile, 8 waves (4x2), each wave 2x2 16x16 accumulators ->
// 4x V_WMMA_F32_16X16X32_BF16 per k-step.  A tile is software-pipelined:
// GLOBAL_LOAD_ASYNC_TO_LDS_B128 for tile kt+1 streams into the other LDS
// buffer while tile kt's WMMAs run; S_WAIT_ASYNCCNT <= 2 (in-order counter)
// releases tile kt.  B tile converted f32->bf16 through ds_store_b16.
// align uses hardware v_tanh_f32 / v_exp_f32.
// ---------------------------------------------------------------------------

#define Bc 8
#define Tc 32
#define Lc 512
#define Dc 1024

typedef __attribute__((ext_vector_type(16))) __bf16 v16bf;
typedef __attribute__((ext_vector_type(8)))  __bf16 v8bf;
typedef __attribute__((ext_vector_type(4)))  __bf16 v4bf;
typedef __attribute__((ext_vector_type(8)))  float  v8f;

__device__ inline __bf16 f2bf(float f) {
    union { float f; unsigned u; } in; in.f = f;
    unsigned r = in.u + 0x7FFFu + ((in.u >> 16) & 1u);   // round-to-nearest-even
    union { unsigned short s; __bf16 b; } out;
    out.s = (unsigned short)(r >> 16);
    return out.b;
}

__device__ inline v16bf mk16(v8bf lo, v8bf hi) {
    return __builtin_shufflevector(lo, hi, 0,1,2,3,4,5,6,7,8,9,10,11,12,13,14,15);
}

__device__ inline float fast_tanh(float x) {
    float y;
    asm volatile("v_tanh_f32 %0, %1\n\tv_nop" : "=v"(y) : "v"(x));
    return y;
}

__device__ inline float fast_exp(float x) {
    return __builtin_amdgcn_exp2f(x * 1.4426950408889634f);  // v_exp_f32
}

template <int N>
__device__ inline void wait_async() {
#if __has_builtin(__builtin_amdgcn_s_wait_asynccnt)
    __builtin_amdgcn_s_wait_asynccnt(N);
#else
    asm volatile("s_wait_asynccnt %0" :: "i"(N) : "memory");
#endif
}

// ---------------------------------------------------------------------------
// k0: f32 -> bf16 copy (n multiple of 4)
// ---------------------------------------------------------------------------
__global__ __launch_bounds__(256)
void cvt_f32_bf16(const float* __restrict__ src, __bf16* __restrict__ dst, long n)
{
    const long i = ((long)blockIdx.x * 256 + threadIdx.x) * 4;
    if (i < n) {
        float4 f = *(const float4*)(src + i);
        v4bf p; p[0]=f2bf(f.x); p[1]=f2bf(f.y); p[2]=f2bf(f.z); p[3]=f2bf(f.w);
        *(v4bf*)(dst + i) = p;
    }
}

// ---------------------------------------------------------------------------
// Generic tiled GEMM:  C[M,N] = A[M,K](bf16) @ B[K,N](f32->bf16)
//   (+bias[n]) (+rowv[m]*colv[n]) (relu) ; C written as f32 or bf16.
// A optionally split at k=splitK between A0/A1.  Batched via grid.z.
// Block = 256 threads = 8 waves; tile 128x64; K stepped by 32.
// A tile double-buffered in LDS, filled by async copies one tile ahead.
// ---------------------------------------------------------------------------
#define SA 40   // LDS row stride (halves) for A tile (128 x 32)
#define SB 40   // LDS col stride (halves) for B tile (64 cols x 32 k, k-contig)

__global__ __launch_bounds__(256)
void gemm_bf16_wmma(const __bf16* __restrict__ A0, const __bf16* __restrict__ A1,
                    int splitK,
                    const float* __restrict__ Bm, void* __restrict__ C, int outBf,
                    const float* __restrict__ bias,
                    const float* __restrict__ rowv, const float* __restrict__ colv,
                    int M, int N, int K, int ldA, int relu,
                    long batchA, long batchB, long batchC)
{
    __shared__ __attribute__((aligned(16))) __bf16 lds_a[2][128 * SA];
    __shared__ __attribute__((aligned(16))) __bf16 lds_b[64 * SB];

    const int tid  = threadIdx.x;
    const int lane = tid & 31;
    const int w    = tid >> 5;          // wave 0..7
    const int mw   = w >> 1;            // 0..3 : wave row  (2 m-subtiles each)
    const int nw   = w & 1;             // 0..1 : wave col  (2 n-subtiles each)

    const long zb = blockIdx.z;
    const __bf16* Ab0 = A0 + zb * batchA;
    const __bf16* Ab1 = A1 ? (A1 + zb * batchA) : nullptr;
    const float*  Bb  = Bm + zb * batchB;
    float*  Cf = (float*)C  + zb * batchC;
    __bf16* Ch = (__bf16*)C + zb * batchC;

    const int m_base = blockIdx.y * 128;
    const int n_base = blockIdx.x * 64;

    // issue async copies for one 128x32 A tile into abuf (2 x b128 per thread)
    auto issue_A = [&](int k0, __bf16* abuf) {
        #pragma unroll
        for (int u = 0; u < 2; ++u) {
            const int id = tid + u * 256;       // 0..511
            const int r  = id >> 2;             // 0..127
            const int c0 = (id & 3) << 3;       // halves 0,8,16,24
            int gm = m_base + r; if (gm >= M) gm = M - 1;   // clamp, unused rows
            const __bf16* gp;
            if (Ab1 && k0 >= splitK) gp = Ab1 + (long)gm * ldA + (k0 - splitK) + c0;
            else                     gp = Ab0 + (long)gm * ldA + k0 + c0;
            const unsigned lds_off = (unsigned)(uintptr_t)(abuf + r * SA + c0);
            const unsigned long long ga = (unsigned long long)(uintptr_t)gp;
            asm volatile("global_load_async_to_lds_b128 %0, %1, off"
                         :: "v"(lds_off), "v"(ga) : "memory");
        }
    };

    v8f acc[2][2] = {};

    const int KT = K >> 5;
    issue_A(0, lds_a[0]);                       // prologue: tile 0 in flight

    for (int kt = 0; kt < KT; ++kt) {
        const int k0 = kt << 5;
        __bf16* acur = lds_a[kt & 1];

        // ---- issue async copies for tile kt+1 into the other buffer ----
        const bool more = (kt + 1 < KT);
        if (more) issue_A(k0 + 32, lds_a[(kt + 1) & 1]);

        // ---- stage B tile (32 k x 64 cols): f32 load, cvt, transposed store ----
        {
            const int kr = tid >> 3;            // 0..31
            const int c0 = (tid & 7) << 3;      // 0..56 step 8
            const float* Brow = Bb + (long)(k0 + kr) * N + n_base + c0;
            float4 f0 = *(const float4*)(Brow);
            float4 f1 = *(const float4*)(Brow + 4);
            lds_b[(c0 + 0) * SB + kr] = f2bf(f0.x);
            lds_b[(c0 + 1) * SB + kr] = f2bf(f0.y);
            lds_b[(c0 + 2) * SB + kr] = f2bf(f0.z);
            lds_b[(c0 + 3) * SB + kr] = f2bf(f0.w);
            lds_b[(c0 + 4) * SB + kr] = f2bf(f1.x);
            lds_b[(c0 + 5) * SB + kr] = f2bf(f1.y);
            lds_b[(c0 + 6) * SB + kr] = f2bf(f1.z);
            lds_b[(c0 + 7) * SB + kr] = f2bf(f1.w);
            if (more)
                __builtin_prefetch(Brow + 32L * N, 0, 1);   // global_prefetch_b8
        }

        // ---- release tile kt: the 2 newer asyncs (tile kt+1) may remain ----
        if (more) wait_async<2>();
        else      wait_async<0>();
        __syncthreads();        // all waves' tile-kt staging visible

        // ---- fragments ----
        // A 16x32 bf16 wave32 layout:
        //   lanes 0-15:  M=lane,     K 0-7 (v0-3) / 16-23 (v4-7)
        //   lanes 16-31: M=lane-16,  K 8-15      / 24-31
        const int h8   = (lane >> 4) << 3;                 // 0 or 8
        const int koff = (lane >> 4) << 4;                 // 0 or 16
        v16bf afrag[2], bfrag[2];
        #pragma unroll
        for (int mi = 0; mi < 2; ++mi) {
            const int arow = (mw * 2 + mi) * 16 + (lane & 15);
            v8bf alo = *(const v8bf*)(&acur[arow * SA + h8]);
            v8bf ahi = *(const v8bf*)(&acur[arow * SA + 16 + h8]);
            afrag[mi] = mk16(alo, ahi);
        }
        // B 32x16 bf16: lane = col; K 0-7/16-23 (lanes 0-15) / 8-15/24-31 (lanes 16-31)
        #pragma unroll
        for (int nj = 0; nj < 2; ++nj) {
            const int col = (nw * 2 + nj) * 16 + (lane & 15);
            v8bf b0 = *(const v8bf*)(&lds_b[col * SB + h8]);
            v8bf b1 = *(const v8bf*)(&lds_b[col * SB + 16 + h8]);
            bfrag[nj] = mk16(b0, b1);
        }

        #pragma unroll
        for (int mi = 0; mi < 2; ++mi)
            #pragma unroll
            for (int nj = 0; nj < 2; ++nj)
                acc[mi][nj] = __builtin_amdgcn_wmma_f32_16x16x32_bf16(
                    false, afrag[mi], false, bfrag[nj], (short)0,
                    acc[mi][nj], false, false);

        __syncthreads();        // seal buffers before next iteration reuses them
    }

    // ---- epilogue: C/D layout — VGPR r: M = r (lanes 0-15), M = 8+r (16-31) ----
    const int rofs = (lane >> 4) << 3;
    #pragma unroll
    for (int mi = 0; mi < 2; ++mi) {
        #pragma unroll
        for (int nj = 0; nj < 2; ++nj) {
            const int col = n_base + (nw * 2 + nj) * 16 + (lane & 15);
            #pragma unroll
            for (int r = 0; r < 8; ++r) {
                const int row = m_base + (mw * 2 + mi) * 16 + rofs + r;
                if (row < M) {
                    float val = acc[mi][nj][r];
                    if (bias) val += bias[col];
                    if (rowv) val += rowv[row] * colv[col];
                    if (relu) val = fmaxf(val, 0.0f);
                    if (outBf) Ch[(long)row * N + col] = f2bf(val);
                    else       Cf[(long)row * N + col] = val;
                }
            }
        }
    }
}

// ---------------------------------------------------------------------------
// align[l] = v . tanh(qf[b,t,:] + sc[b,l,:]);  mask;  softmax over L;
// attn (B,T,L) -> ws as bf16 (A-operand of k4); attn_t / new_coverage -> d_out.
// One block per (b,t); 8 waves; each wave owns 64 l-rows; lanes parallel
// over D (coalesced 512B per wave per step) with cross-lane reduction.
// ---------------------------------------------------------------------------
__global__ __launch_bounds__(256)
void align_softmax(const float* __restrict__ qf, const float* __restrict__ sc,
                   const float* __restrict__ vvec,
                   const unsigned char* __restrict__ mask,
                   const float* __restrict__ coverage,
                   __bf16* __restrict__ attn_bf,
                   float* __restrict__ out_newcov, float* __restrict__ out_attn_t)
{
    const int bt  = blockIdx.x;         // b*T + t
    const int b   = bt >> 5;            // T = 32
    const int tt  = bt & 31;
    const int tid = threadIdx.x;
    const int lane = tid & 31;
    const int w    = tid >> 5;

    __shared__ float s_q[Dc];
    __shared__ float s_v[Dc];
    __shared__ float s_a[Lc];
    __shared__ float s_r[Lc];

    for (int i = tid; i < Dc; i += 256) {
        s_q[i] = qf[(long)bt * Dc + i];
        s_v[i] = vvec[i];
    }
    __syncthreads();

    for (int lw = 0; lw < 64; ++lw) {
        const int l = w * 64 + lw;
        const float* srow = sc + ((long)b * Lc + l) * Dc;
        float acc = 0.0f;
        const int d0 = lane << 2;
        #pragma unroll
        for (int i = 0; i < Dc / 128; ++i) {
            const int d = d0 + i * 128;
            float4 s4 = *(const float4*)(srow + d);
            acc += s_v[d + 0] * fast_tanh(s_q[d + 0] + s4.x);
            acc += s_v[d + 1] * fast_tanh(s_q[d + 1] + s4.y);
            acc += s_v[d + 2] * fast_tanh(s_q[d + 2] + s4.z);
            acc += s_v[d + 3] * fast_tanh(s_q[d + 3] + s4.w);
        }
        #pragma unroll
        for (int off = 16; off > 0; off >>= 1)
            acc += __shfl_down(acc, off, 32);
        if (lane == 0) {
            if (!mask[b * Lc + l]) acc = -__builtin_inff();
            s_a[l] = acc;
        }
    }
    __syncthreads();

    // softmax over L = 512 with 256 threads
    s_r[tid] = fmaxf(s_a[tid], s_a[tid + 256]);
    __syncthreads();
    for (int s = 128; s > 0; s >>= 1) {
        if (tid < s) s_r[tid] = fmaxf(s_r[tid], s_r[tid + s]);
        __syncthreads();
    }
    const float mx = s_r[0];
    __syncthreads();

    const float e0 = fast_exp(s_a[tid * 2]     - mx);
    const float e1 = fast_exp(s_a[tid * 2 + 1] - mx);
    s_a[tid * 2]     = e0;
    s_a[tid * 2 + 1] = e1;
    s_r[tid] = e0 + e1;
    __syncthreads();
    for (int s = 128; s > 0; s >>= 1) {
        if (tid < s) s_r[tid] += s_r[tid + s];
        __syncthreads();
    }
    const float inv = 1.0f / s_r[0];

    #pragma unroll
    for (int li = 0; li < 2; ++li) {
        const int l = tid * 2 + li;
        const float a = s_a[l] * inv;
        attn_bf[(long)bt * Lc + l] = f2bf(a);
        const long o = ((long)b * Lc + l) * Tc + tt;
        out_attn_t[o] = a;
        out_newcov[o] = coverage[b * Lc + l] + a;
    }
}

// ---------------------------------------------------------------------------
extern "C" void kernel_launch(void* const* d_in, const int* in_sizes, int n_in,
                              void* d_out, int out_size, void* d_ws, size_t ws_size,
                              hipStream_t stream)
{
    const float* query  = (const float*)d_in[0];   // (B,T,D)
    const float* states = (const float*)d_in[1];   // (B,L,D)
    const unsigned char* mask = (const unsigned char*)d_in[2]; // (B,L) bool
    const float* cov    = (const float*)d_in[3];   // (B,L,1)
    const float* Wq     = (const float*)d_in[4];   // (D,D)
    const float* bq     = (const float*)d_in[5];   // (D)
    const float* Ws     = (const float*)d_in[6];   // (D,D)
    const float* Wcov   = (const float*)d_in[7];   // (1,D)
    const float* vvec   = (const float*)d_in[8];   // (D)
    const float* W1     = (const float*)d_in[9];   // (2D,D)
    const float* b1     = (const float*)d_in[10];  // (D)
    const float* W2     = (const float*)d_in[11];  // (D,D)
    const float* b2     = (const float*)d_in[12];  // (D)

    float* out_h    = (float*)d_out;                       // (B,T,D)   262144
    float* out_ncov = (float*)d_out + (long)Bc*Tc*Dc;      // (B,L,T)   131072
    float* out_attT = out_ncov + (long)Bc*Lc*Tc;           // (B,L,T)   131072

    // workspace layout (~27 MB)
    char* wsb = (char*)d_ws;
    float*  qf     = (float*)wsb;                 wsb += (long)Bc*Tc*Dc*4;   // 1 MB
    float*  sc     = (float*)wsb;                 wsb += (long)Bc*Lc*Dc*4;   // 16 MB
    __bf16* qbf    = (__bf16*)wsb;                wsb += (long)Bc*Tc*Dc*2;   // .5 MB
    __bf16* sbf    = (__bf16*)wsb;                wsb += (long)Bc*Lc*Dc*2;   // 8 MB
    __bf16* attnbf = (__bf16*)wsb;                wsb += (long)Bc*Tc*Lc*2;   // .25 MB
    __bf16* ctxbf  = (__bf16*)wsb;                wsb += (long)Bc*Tc*Dc*2;   // .5 MB
    __bf16* h1bf   = (__bf16*)wsb;                wsb += (long)Bc*Tc*Dc*2;   // .5 MB

    dim3 blk(256);

    // k0: bf16 copies of the A-operands that come from inputs
    cvt_f32_bf16<<<dim3(((long)Bc*Tc*Dc)/1024), blk, 0, stream>>>(query,  qbf, (long)Bc*Tc*Dc);
    cvt_f32_bf16<<<dim3(((long)Bc*Lc*Dc)/1024), blk, 0, stream>>>(states, sbf, (long)Bc*Lc*Dc);

    // k1: qf = query @ Wq + bq   (M=256, N=1024, K=1024) -> f32
    gemm_bf16_wmma<<<dim3(16, 2, 1), blk, 0, stream>>>(
        qbf, nullptr, 0, Wq, qf, 0, bq, nullptr, nullptr,
        Bc*Tc, Dc, Dc, Dc, 0, 0, 0, 0);

    // k2: sc = states @ Ws + coverage x Wcov   (M=4096, N=1024, K=1024) -> f32
    gemm_bf16_wmma<<<dim3(16, 32, 1), blk, 0, stream>>>(
        sbf, nullptr, 0, Ws, sc, 0, nullptr, cov, Wcov,
        Bc*Lc, Dc, Dc, Dc, 0, 0, 0, 0);

    // k3: align + softmax + attn_t + new_coverage; attn -> bf16
    align_softmax<<<dim3(Bc*Tc), blk, 0, stream>>>(
        qf, sc, vvec, mask, cov, attnbf, out_ncov, out_attT);

    // k4: ctx[b] = attn[b] @ states[b]   (batched: M=32, N=1024, K=512) -> bf16
    gemm_bf16_wmma<<<dim3(16, 1, Bc), blk, 0, stream>>>(
        attnbf, nullptr, 0, states, ctxbf, 1, nullptr, nullptr, nullptr,
        Tc, Dc, Lc, Lc, 0,
        (long)Tc*Lc, (long)Lc*Dc, (long)Tc*Dc);

    // k5: h1 = relu([ctx, query] @ W1 + b1)  (M=256, N=1024, K=2048, split A) -> bf16
    gemm_bf16_wmma<<<dim3(16, 2, 1), blk, 0, stream>>>(
        ctxbf, qbf, Dc, W1, h1bf, 1, b1, nullptr, nullptr,
        Bc*Tc, Dc, 2*Dc, Dc, 1, 0, 0, 0);

    // k6: h = h1 @ W2 + b2   (M=256, N=1024, K=1024) -> f32 (d_out)
    gemm_bf16_wmma<<<dim3(16, 2, 1), blk, 0, stream>>>(
        h1bf, nullptr, 0, W2, out_h, 0, b2, nullptr, nullptr,
        Bc*Tc, Dc, Dc, Dc, 0, 0, 0, 0);
}
